// MultiheadDiffAttn_16982300688399
// MI455X (gfx1250) — matrix operand order repaired and checked by reading, not compiled, not run on hardware
//
#include <hip/hip_runtime.h>
#include <hip/hip_bf16.h>

// ---------------------------------------------------------------------------
// Multihead Differential Attention (DIFF Transformer style) for MI455X/gfx1250
// bf16 WMMA everywhere: v_wmma_f32_16x16x32_bf16 (K=32 == head_dim)
// ---------------------------------------------------------------------------

typedef __attribute__((ext_vector_type(16))) __bf16 v16bf;
typedef __attribute__((ext_vector_type(8)))  __bf16 v8bf;
typedef __attribute__((ext_vector_type(8)))  float  v8f;
typedef int v2i_ __attribute__((vector_size(8)));   // b64 payload type

#define AS1 __attribute__((address_space(1)))
#define AS3 __attribute__((address_space(3)))

#define LAMBDA_INIT 0.7836057665f   // 0.8 - 0.6*exp(-0.3*12)
#define ONE_MINUS_LI 0.2163942335f
#define ATTN_SCALE   0.1767766953f  // 32^-0.5

#define E_   1024
#define T_   1024
#define B_   4
#define BT_  4096
#define H_   16
#define KVH_ 8
#define HD_  32

// CDNA5 async global->LDS path (ASYNCcnt-tracked), probe-guarded.
#if defined(__AMDGCN__) && __has_builtin(__builtin_amdgcn_global_load_async_to_lds_b64)
#define HAVE_ASYNC_LDS 1
#endif

__device__ inline v8f wmma_bf16(v16bf a, v16bf b, v8f c) {
  // 8 args: (neg_a, A, neg_b, B, c_mod, C, reuse_a, reuse_b)
  return __builtin_amdgcn_wmma_f32_16x16x32_bf16(false, a, false, b, (short)0,
                                                 c, false, false);
}

// Build a v16bf fragment from two 16-byte chunks.
__device__ inline v16bf ld_frag(const __bf16* p0, const __bf16* p1) {
  union { v16bf v; v8bf h[2]; } u;
  u.h[0] = *(const v8bf*)p0;
  u.h[1] = *(const v8bf*)p1;
  return u.v;
}

#if defined(HAVE_ASYNC_LDS)
// global -> LDS 8-byte async copy (per lane), ASYNCcnt-tracked.
__device__ inline void async_b64(const void* g, void* lds) {
  __builtin_amdgcn_global_load_async_to_lds_b64(
      (AS1 v2i_*)(void*)g, (AS3 v2i_*)lds, 0, 0);
}
#endif

__device__ inline void wait_async0() {
#if defined(HAVE_ASYNC_LDS)
#if __has_builtin(__builtin_amdgcn_s_wait_asynccnt)
  __builtin_amdgcn_s_wait_asynccnt(0);
#else
  asm volatile("s_wait_asynccnt 0x0" ::: "memory");
#endif
#endif
}

// ---------------------------------------------------------------------------
// Elementwise converters
// ---------------------------------------------------------------------------
__global__ void k_cvt(const float* __restrict__ s, __bf16* __restrict__ d, int n) {
  int i = blockIdx.x * 256 + threadIdx.x;
  if (i < n) d[i] = (__bf16)s[i];
}

// W (K x N, row-major fp32) -> WT (N x K, row-major bf16)
__global__ void k_cvt_t(const float* __restrict__ W, __bf16* __restrict__ WT,
                        int K, int N) {
  int i = blockIdx.x * 256 + threadIdx.x;
  if (i >= K * N) return;
  int k = i / N, n = i - k * N;
  WT[(size_t)n * K + k] = (__bf16)W[i];
}

// ---------------------------------------------------------------------------
// Generic WMMA GEMM: C(fp32, MxN) = A(bf16, MxK, row-major) @ B(bf16, KxN)
// where B is supplied transposed: BT is N x K row-major.
// Block = 256 thr (8 waves); block tile 256(M) x 64(N); wave tile 32 x 64
// (two A-fragments share each B-fragment -> 8 WMMAs per 12 b128 loads).
// ---------------------------------------------------------------------------
__global__ __launch_bounds__(256) void k_gemm(const __bf16* __restrict__ A,
                                              const __bf16* __restrict__ BT,
                                              float* __restrict__ C,
                                              int M, int N, int K) {
  const int lane = threadIdx.x & 31;
  const int wave = threadIdx.x >> 5;
  const int r  = lane & 15;       // A-frag row / B-frag col / C col
  const int hi = lane >> 4;       // K-half selector
  const int m0 = blockIdx.y * 256 + wave * 32;
  const int n0 = blockIdx.x * 64;

  v8f acc[2][4] = {};
  const __bf16* arow0 = A + (size_t)(m0 + r) * K + hi * 8;
  const __bf16* arow1 = A + (size_t)(m0 + 16 + r) * K + hi * 8;

  for (int k0 = 0; k0 < K; k0 += 32) {
    const __bf16* pa0 = arow0 + k0;
    const __bf16* pa1 = arow1 + k0;
    __builtin_prefetch(pa0 + 64, 0, 3);         // WGP-scope global_prefetch_b8
    __builtin_prefetch(pa1 + 64, 0, 3);
    v16bf a0 = ld_frag(pa0, pa0 + 16);          // A 16x32, ISA layout
    v16bf a1 = ld_frag(pa1, pa1 + 16);
#pragma unroll
    for (int nt = 0; nt < 4; ++nt) {
      const __bf16* pb = BT + (size_t)(n0 + nt * 16 + r) * K + k0 + hi * 8;
      __builtin_prefetch(pb + 32, 0, 3);
      v16bf bf_ = ld_frag(pb, pb + 16);         // B 32x16, ISA layout
      acc[0][nt] = wmma_bf16(a0, bf_, acc[0][nt]);
      acc[1][nt] = wmma_bf16(a1, bf_, acc[1][nt]);
    }
  }
  // C/D layout: VGPR v, lanes 0-15 -> M=v, lanes 16-31 -> M=v+8; N=lane&15
#pragma unroll
  for (int g = 0; g < 2; ++g)
#pragma unroll
    for (int nt = 0; nt < 4; ++nt)
#pragma unroll
      for (int v = 0; v < 8; ++v)
        C[(size_t)(m0 + g * 16 + v + 8 * hi) * N + n0 + nt * 16 + r] =
            acc[g][nt][v];
}

// ---------------------------------------------------------------------------
// RoPE + stream split: src fp32 (BT, nh2, 32) -> out1/out2 bf16 (BT, nh2/2, 32)
// even heads -> stream 1, odd heads -> stream 2. Pair (i, i+16), i in [0,16).
// ---------------------------------------------------------------------------
__global__ void k_rope(const float* __restrict__ src, __bf16* __restrict__ out1,
                       __bf16* __restrict__ out2, int nh2) {
  int idx = blockIdx.x * 256 + threadIdx.x;
  int total = BT_ * nh2 * 16;
  if (idx >= total) return;
  int i  = idx & 15;
  int h2 = (idx >> 4) % nh2;
  int tt = (idx >> 4) / nh2;       // flat b*T + t
  int pos = tt & (T_ - 1);
  float fr = (float)pos * __powf(10000.0f, -(float)i * (1.0f / 16.0f));
  float s, c;
  __sincosf(fr, &s, &c);
  const float* p = src + ((size_t)tt * nh2 + h2) * HD_;
  float x1 = p[i], x2 = p[i + 16];
  float y1 =  x1 * c + x2 * s;
  float y2 = -x1 * s + x2 * c;
  int nh = nh2 >> 1;
  __bf16* o = ((h2 & 1) ? out2 : out1) + ((size_t)tt * nh + (h2 >> 1)) * HD_;
  o[i]      = (__bf16)y1;
  o[i + 16] = (__bf16)y2;
}

// ---------------------------------------------------------------------------
// Fused differential flash attention.
// grid: (B*H, T/128). block: 256 thr = 8 waves; wave owns 16 query rows.
// ---------------------------------------------------------------------------
#define TQ 128
#define TKV 32

__global__ __launch_bounds__(256) void k_attn(
    const __bf16* __restrict__ q1, const __bf16* __restrict__ q2,   // (BT,H,32)
    const __bf16* __restrict__ k1, const __bf16* __restrict__ k2,   // (BT,KVH,32)
    const __bf16* __restrict__ vh,                                  // (BT,KVH,64)
    const float* __restrict__ lq1, const float* __restrict__ lk1,
    const float* __restrict__ lq2, const float* __restrict__ lk2,
    const float* __restrict__ rmsw,
    __bf16* __restrict__ ah)                                        // (BT,H,64)
{
  __shared__ __align__(32) __bf16 ldsK1[TKV * 32];
  __shared__ __align__(32) __bf16 ldsK2[TKV * 32];
  __shared__ __align__(32) __bf16 ldsVT[64 * TKV];      // transposed: [d][key]
  __shared__ __align__(32) __bf16 ldsP[8][2][16 * TKV]; // per-wave P scratch

  const int lane = threadIdx.x & 31;
  const int wave = threadIdx.x >> 5;
  const int r  = lane & 15;
  const int hi = lane >> 4;
  const int bh = blockIdx.x;
  const int qb = blockIdx.y;
  const int b = bh / H_, h = bh - b * H_, kvh = h >> 1;
  const int t0 = qb * TQ;

  // lambda_full = exp(sum lq1*lk1) - exp(sum lq2*lk2) + LAMBDA_INIT  (HD=32 lanes)
  float lam;
  {
    float p1 = lq1[lane] * lk1[lane];
    float p2 = lq2[lane] * lk2[lane];
#pragma unroll
    for (int m = 16; m >= 1; m >>= 1) {
      p1 += __shfl_xor(p1, m, 32);
      p2 += __shfl_xor(p2, m, 32);
    }
    lam = __expf(p1) - __expf(p2) + LAMBDA_INIT;
  }

  // Q fragments (constant across the KV loop)
  const int tq = t0 + wave * 16 + r;
  const __bf16* pq1 = q1 + ((size_t)(b * T_ + tq) * H_ + h) * HD_ + hi * 8;
  const __bf16* pq2 = q2 + ((size_t)(b * T_ + tq) * H_ + h) * HD_ + hi * 8;
  v16bf aq1 = ld_frag(pq1, pq1 + 16);
  v16bf aq2 = ld_frag(pq2, pq2 + 16);

  float m1[8], l1[8], m2[8], l2[8];
  v8f o1[4] = {}, o2[4] = {};
#pragma unroll
  for (int v = 0; v < 8; ++v) { m1[v] = -1e30f; m2[v] = -1e30f; l1[v] = 0.f; l2[v] = 0.f; }

  const int ntiles = (qb + 1) * (TQ / TKV);   // causal extent

  for (int it = 0; it < ntiles; ++it) {
    const int kv0 = it * TKV;
    __syncthreads();  // previous iteration finished reading LDS
    // ---- stage K1/K2 (32 keys x 32) and V^T (64 x 32) ----
    {
      int key = threadIdx.x >> 3;
      int j4  = (threadIdx.x & 7) * 4;
      const __bf16* g1 = k1 + ((size_t)(b * T_ + kv0 + key) * KVH_ + kvh) * HD_ + j4;
      const __bf16* g2 = k2 + ((size_t)(b * T_ + kv0 + key) * KVH_ + kvh) * HD_ + j4;
#if defined(HAVE_ASYNC_LDS)
      // CDNA5 async data path: global -> LDS without VGPR round trip.
      async_b64(g1, &ldsK1[key * 32 + j4]);
      async_b64(g2, &ldsK2[key * 32 + j4]);
#else
#pragma unroll
      for (int i = 0; i < 4; ++i) {
        ldsK1[key * 32 + j4 + i] = g1[i];
        ldsK2[key * 32 + j4 + i] = g2[i];
      }
#endif
      int j8 = (threadIdx.x & 7) * 8;
      const __bf16* gv = vh + ((size_t)(b * T_ + kv0 + key) * KVH_ + kvh) * 64 + j8;
      v8bf vv = *(const v8bf*)gv;               // one b128 load
#pragma unroll
      for (int i = 0; i < 8; ++i) ldsVT[(j8 + i) * TKV + key] = vv[i];
    }
    wait_async0();
    __syncthreads();

    // ---- scores: S = Q K^T (16x32 chunk = 2 WMMAs per stream) ----
    v8f s1[2], s2[2];
#pragma unroll
    for (int nt = 0; nt < 2; ++nt) {
      const __bf16* pk1 = ldsK1 + (nt * 16 + r) * 32 + hi * 8;
      const __bf16* pk2 = ldsK2 + (nt * 16 + r) * 32 + hi * 8;
      v16bf bk1 = ld_frag(pk1, pk1 + 16);
      v16bf bk2 = ld_frag(pk2, pk2 + 16);
      v8f z = {};
      s1[nt] = wmma_bf16(aq1, bk1, z);
      s2[nt] = wmma_bf16(aq2, bk2, z);
    }

    // ---- scale + causal mask + online softmax (both streams) ----
#pragma unroll
    for (int v = 0; v < 8; ++v) {
      const int trow = t0 + wave * 16 + v + 8 * hi;
#pragma unroll
      for (int nt = 0; nt < 2; ++nt) {
        const int key = kv0 + nt * 16 + r;
        float x1 = s1[nt][v] * ATTN_SCALE;
        float x2 = s2[nt][v] * ATTN_SCALE;
        if (key > trow) { x1 = -1e30f; x2 = -1e30f; }
        s1[nt][v] = x1; s2[nt][v] = x2;
      }
      float a_ = fmaxf(s1[0][v], s1[1][v]);
      float b_ = fmaxf(s2[0][v], s2[1][v]);
#pragma unroll
      for (int m = 8; m >= 1; m >>= 1) {   // reduce within 16-lane half
        a_ = fmaxf(a_, __shfl_xor(a_, m, 32));
        b_ = fmaxf(b_, __shfl_xor(b_, m, 32));
      }
      float mn1 = fmaxf(m1[v], a_);
      float mn2 = fmaxf(m2[v], b_);
      float c1 = __expf(m1[v] - mn1);
      float c2 = __expf(m2[v] - mn2);
      float p10 = __expf(s1[0][v] - mn1), p11 = __expf(s1[1][v] - mn1);
      float p20 = __expf(s2[0][v] - mn2), p21 = __expf(s2[1][v] - mn2);
      float rs1 = p10 + p11, rs2 = p20 + p21;
#pragma unroll
      for (int m = 8; m >= 1; m >>= 1) {
        rs1 += __shfl_xor(rs1, m, 32);
        rs2 += __shfl_xor(rs2, m, 32);
      }
      l1[v] = l1[v] * c1 + rs1;  m1[v] = mn1;
      l2[v] = l2[v] * c2 + rs2;  m2[v] = mn2;
#pragma unroll
      for (int nt = 0; nt < 4; ++nt) { o1[nt][v] *= c1; o2[nt][v] *= c2; }
      const int row = v + 8 * hi;
      ldsP[wave][0][row * 32 +      r] = (__bf16)p10;
      ldsP[wave][0][row * 32 + 16 + r] = (__bf16)p11;
      ldsP[wave][1][row * 32 +      r] = (__bf16)p20;
      ldsP[wave][1][row * 32 + 16 + r] = (__bf16)p21;
    }
    __syncthreads();   // P re-fragmentation through LDS

    // ---- O += P @ V (4 output tiles x 2 streams) ----
    const __bf16* pp1 = &ldsP[wave][0][r * 32 + hi * 8];
    const __bf16* pp2 = &ldsP[wave][1][r * 32 + hi * 8];
    v16bf ap1 = ld_frag(pp1, pp1 + 16);
    v16bf ap2 = ld_frag(pp2, pp2 + 16);
#pragma unroll
    for (int nt = 0; nt < 4; ++nt) {
      const __bf16* pv = ldsVT + (nt * 16 + r) * TKV + hi * 8;
      v16bf bv = ld_frag(pv, pv + 16);
      o1[nt] = wmma_bf16(ap1, bv, o1[nt]);
      o2[nt] = wmma_bf16(ap2, bv, o2[nt]);
    }
  }

  // ---- epilogue: combine streams, RMS-norm over 64, scale, store bf16 ----
  float comb[4][8];
  float sumsq[8];
#pragma unroll
  for (int v = 0; v < 8; ++v) sumsq[v] = 0.f;
#pragma unroll
  for (int nt = 0; nt < 4; ++nt)
#pragma unroll
    for (int v = 0; v < 8; ++v) {
      float a1 = o1[nt][v] / l1[v];
      float a2 = o2[nt][v] / l2[v];
      float c = a1 - lam * a2;
      comb[nt][v] = c;
      sumsq[v] += c * c;
    }
#pragma unroll
  for (int v = 0; v < 8; ++v) {
#pragma unroll
    for (int m = 8; m >= 1; m >>= 1) sumsq[v] += __shfl_xor(sumsq[v], m, 32);
    sumsq[v] = rsqrtf(sumsq[v] * (1.0f / 64.0f) + 1e-6f);
  }
#pragma unroll
  for (int nt = 0; nt < 4; ++nt) {
    float w = rmsw[nt * 16 + r] * ONE_MINUS_LI;
#pragma unroll
    for (int v = 0; v < 8; ++v) {
      int trow = t0 + wave * 16 + v + 8 * hi;
      float outv = comb[nt][v] * sumsq[v] * w;
      ah[((size_t)(b * T_ + trow) * H_ + h) * 64 + nt * 16 + r] = (__bf16)outv;
    }
  }
}

// ---------------------------------------------------------------------------
// Host-side orchestration
// ---------------------------------------------------------------------------
extern "C" void kernel_launch(void* const* d_in, const int* in_sizes, int n_in,
                              void* d_out, int out_size, void* d_ws, size_t ws_size,
                              hipStream_t stream) {
  (void)in_sizes; (void)n_in; (void)out_size; (void)ws_size;
  const float* x    = (const float*)d_in[0];
  const float* Wq   = (const float*)d_in[1];
  const float* Wk   = (const float*)d_in[2];
  const float* Wv   = (const float*)d_in[3];
  const float* Wo   = (const float*)d_in[4];
  const float* lq1  = (const float*)d_in[5];
  const float* lk1  = (const float*)d_in[6];
  const float* lq2  = (const float*)d_in[7];
  const float* lk2  = (const float*)d_in[8];
  const float* rmsw = (const float*)d_in[9];
  float* out = (float*)d_out;

  char* ws = (char*)d_ws;
  size_t off = 0;
  auto take = [&](size_t bytes) -> void* {
    void* p = ws + off;
    off += (bytes + 255) & ~(size_t)255;
    return p;
  };
  __bf16* xh  = (__bf16*)take((size_t)BT_ * E_ * 2);
  __bf16* WqT = (__bf16*)take((size_t)E_ * E_ * 2);
  __bf16* WkT = (__bf16*)take((size_t)E_ * 512 * 2);
  __bf16* WvT = (__bf16*)take((size_t)E_ * 512 * 2);
  __bf16* WoT = (__bf16*)take((size_t)E_ * E_ * 2);
  float*  qf  = (float*)take((size_t)BT_ * E_ * 4);
  float*  kf  = (float*)take((size_t)BT_ * 512 * 4);
  float*  vf  = (float*)take((size_t)BT_ * 512 * 4);
  __bf16* q1h = (__bf16*)take((size_t)BT_ * H_ * HD_ * 2);
  __bf16* q2h = (__bf16*)take((size_t)BT_ * H_ * HD_ * 2);
  __bf16* k1h = (__bf16*)take((size_t)BT_ * KVH_ * HD_ * 2);
  __bf16* k2h = (__bf16*)take((size_t)BT_ * KVH_ * HD_ * 2);
  __bf16* vh  = (__bf16*)take((size_t)BT_ * KVH_ * 64 * 2);
  __bf16* ah  = (__bf16*)take((size_t)BT_ * E_ * 2);

  // 1) convert inputs to bf16 (weights transposed for contiguous B-fragments)
  k_cvt  <<<(BT_ * E_ + 255) / 256, 256, 0, stream>>>(x, xh, BT_ * E_);
  k_cvt_t<<<(E_ * E_  + 255) / 256, 256, 0, stream>>>(Wq, WqT, E_, E_);
  k_cvt_t<<<(E_ * 512 + 255) / 256, 256, 0, stream>>>(Wk, WkT, E_, 512);
  k_cvt_t<<<(E_ * 512 + 255) / 256, 256, 0, stream>>>(Wv, WvT, E_, 512);
  k_cvt_t<<<(E_ * E_  + 255) / 256, 256, 0, stream>>>(Wo, WoT, E_, E_);

  // 2) projections (WMMA)
  k_gemm<<<dim3(E_ / 64,  BT_ / 256), 256, 0, stream>>>(xh, WqT, qf, BT_, E_,  E_);
  k_gemm<<<dim3(512 / 64, BT_ / 256), 256, 0, stream>>>(xh, WkT, kf, BT_, 512, E_);
  k_gemm<<<dim3(512 / 64, BT_ / 256), 256, 0, stream>>>(xh, WvT, vf, BT_, 512, E_);

  // 3) RoPE + stream split + V conversion
  k_rope<<<(BT_ * 2 * H_  * 16 + 255) / 256, 256, 0, stream>>>(qf, q1h, q2h, 2 * H_);
  k_rope<<<(BT_ * 2 * KVH_* 16 + 255) / 256, 256, 0, stream>>>(kf, k1h, k2h, 2 * KVH_);
  k_cvt <<<(BT_ * 512 + 255) / 256, 256, 0, stream>>>(vf, vh, BT_ * 512);

  // 4) fused differential flash attention (WMMA + async LDS staging)
  k_attn<<<dim3(B_ * H_, T_ / TQ), 256, 0, stream>>>(q1h, q2h, k1h, k2h, vh,
                                                     lq1, lk1, lq2, lk2, rmsw, ah);

  // 5) output projection (WMMA) -> fp32 result
  k_gemm<<<dim3(E_ / 64, BT_ / 256), 256, 0, stream>>>(ah, WoT, out, BT_, E_, E_);
}
